// CustomRoFormerSelfAttention_53523882443708
// MI455X (gfx1250) — matrix operand order repaired and checked by reading, not compiled, not run on hardware
//
#include <hip/hip_runtime.h>
#include <hip/hip_bf16.h>

typedef __attribute__((ext_vector_type(16))) _Float16 v16h;
typedef __attribute__((ext_vector_type(8)))  float    v8f;

typedef unsigned int u32x4 __attribute__((ext_vector_type(4)));
typedef int          i32x8 __attribute__((ext_vector_type(8)));
typedef int          i32x4 __attribute__((ext_vector_type(4)));

#define H_DIM 1024
#define S_LEN 2048
#define BATCH 4
#define NH    16
#define HD    64
#define TH3   3072
#define BS    (BATCH * S_LEN)   // 8192

#if defined(__has_builtin)
#  if __has_builtin(__builtin_amdgcn_tensor_load_to_lds)
#    define HAVE_TDM 1
#  endif
#endif
#ifndef HAVE_TDM
#  define HAVE_TDM 0
#endif

struct V16U { union { v16h v; uint4 q[2]; _Float16 h[16]; }; };

__device__ __forceinline__ v8f wmma_f16(v16h a, v16h b, v8f c) {
    // D = A(16x32 f16) * B(32x16 f16) + C(16x16 f32)
    return __builtin_amdgcn_wmma_f32_16x16x32_f16(false, a, false, b, (short)0, c,
                                                  false, false);
}

// A fragment (16x32 f16) from row-major LDS tile (ISA 7.12.2 16-bit A layout).
__device__ __forceinline__ v16h load_a_frag(const _Float16* base, int stride_h, int lane) {
    int m = lane & 15, hs = lane >> 4;
    const _Float16* r = base + m * stride_h;
    V16U t;
    t.q[0] = *(const uint4*)(r + hs * 8);
    t.q[1] = *(const uint4*)(r + 16 + hs * 8);
    return t.v;
}

// B fragment (32x16 f16) from LDS stored as [n][k] (ISA 7.12.2 16-bit B layout).
__device__ __forceinline__ v16h load_b_frag(const _Float16* base, int stride_h, int lane) {
    int n = lane & 15, kh = lane >> 4;
    const _Float16* r = base + n * stride_h + kh * 8;
    V16U t;
    t.q[0] = *(const uint4*)(r);
    t.q[1] = *(const uint4*)(r + 16);
    return t.v;
}

__device__ __forceinline__ unsigned lds_addr_of(const void* p) {
    // generic pointer to LDS: low 32 bits are the wave-relative LDS byte address
    return (unsigned)(unsigned long long)p;
}

#if HAVE_TDM
// Issue a TDM 2D tile load Global->LDS. Element size fixed at 2 bytes (f16).
// pad_interval_code: 0=2,1=4,2=8,3=16,4=32,... DWORDs between pads
// pad_amount_code:   0=1,1=2,2=3,3=4,...       DWORDs of pad inserted
__device__ __forceinline__ void tdm_load_2d_f16(const void* gaddr, unsigned lds_off,
                                                unsigned tensor_d0, unsigned tensor_d1,
                                                unsigned stride0_elems,
                                                unsigned tile_d0, unsigned tile_d1,
                                                unsigned pad_interval_code,
                                                unsigned pad_amount_code) {
    unsigned long long ga = (unsigned long long)gaddr;
    u32x4 g0;
    g0[0] = 1u;                                              // count=1 (valid, user mode)
    g0[1] = lds_off;                                         // lds_addr (bytes)
    g0[2] = (unsigned)(ga & 0xffffffffu);                    // global_addr[31:0]
    g0[3] = (unsigned)((ga >> 32) & 0x1ffffffu) | (2u << 30); // addr[56:32] | type=2
    i32x8 g1;
    g1[0] = (int)((1u << 16) |                               // data_size = 2 bytes
                  (1u << 20) |                               // pad_enable
                  (pad_interval_code << 22) |
                  (pad_amount_code << 25));
    g1[1] = (int)(tensor_d0 << 16);                          // tensor_dim0[15:0]
    g1[2] = (int)((tensor_d0 >> 16) | (tensor_d1 << 16));    // d0[31:16] | d1[15:0]
    g1[3] = (int)((tensor_d1 >> 16) | (tile_d0 << 16));      // d1[31:16] | tile_dim0
    g1[4] = (int)(tile_d1 & 0xffffu);                        // tile_dim1 (tile_dim2=0)
    g1[5] = (int)stride0_elems;                              // tensor_dim0_stride[31:0]
    g1[6] = 0;                                               // stride0[47:32] | stride1[15:0]
    g1[7] = 0;                                               // stride1[47:16]
    i32x4 g2 = {0, 0, 0, 0};
    i32x4 g3 = {0, 0, 0, 0};
    i32x8 g4 = {0, 0, 0, 0, 0, 0, 0, 0};                     // 6-arg toolchain form
    __builtin_amdgcn_tensor_load_to_lds(g0, g1, g2, g3, g4, 0);
}
#endif

// ---------------- kernel 1: w_qkv fp32 -> f16 ----------------
__global__ void cvt_w_kernel(const float* __restrict__ w, _Float16* __restrict__ w16) {
    int i = (blockIdx.x * blockDim.x + threadIdx.x) * 4;
    float4 v = *(const float4*)(w + i);
    w16[i + 0] = (_Float16)v.x;
    w16[i + 1] = (_Float16)v.y;
    w16[i + 2] = (_Float16)v.z;
    w16[i + 3] = (_Float16)v.w;
}

// ---------------- kernel 2: LayerNorm -> f16 ----------------
__global__ __launch_bounds__(256) void ln_kernel(const float* __restrict__ x,
                                                 const float* __restrict__ lw,
                                                 const float* __restrict__ lb,
                                                 _Float16* __restrict__ h16) {
    __shared__ float red[16];
    const int row = blockIdx.x;
    const int tid = threadIdx.x;
    const float* xr = x + (size_t)row * H_DIM;
    float4 v = ((const float4*)xr)[tid];
    float s  = v.x + v.y + v.z + v.w;
    float ss = v.x * v.x + v.y * v.y + v.z * v.z + v.w * v.w;
    #pragma unroll
    for (int off = 16; off > 0; off >>= 1) {
        s  += __shfl_down(s, off, 32);
        ss += __shfl_down(ss, off, 32);
    }
    int wave = tid >> 5, lane = tid & 31;
    if (lane == 0) { red[wave] = s; red[8 + wave] = ss; }
    __syncthreads();
    if (wave == 0) {
        float s2  = (lane < 8) ? red[lane] : 0.0f;
        float ss2 = (lane < 8) ? red[8 + lane] : 0.0f;
        #pragma unroll
        for (int off = 4; off > 0; off >>= 1) {
            s2  += __shfl_down(s2, off, 32);
            ss2 += __shfl_down(ss2, off, 32);
        }
        if (lane == 0) { red[0] = s2; red[1] = ss2; }
    }
    __syncthreads();
    float mu  = red[0] * (1.0f / H_DIM);
    float var = red[1] * (1.0f / H_DIM) - mu * mu;
    float inv = rsqrtf(var + 1e-12f);
    int c = tid * 4;
    _Float16* o = h16 + (size_t)row * H_DIM + c;
    o[0] = (_Float16)(((v.x - mu) * inv) * lw[c + 0] + lb[c + 0]);
    o[1] = (_Float16)(((v.y - mu) * inv) * lw[c + 1] + lb[c + 1]);
    o[2] = (_Float16)(((v.z - mu) * inv) * lw[c + 2] + lb[c + 2]);
    o[3] = (_Float16)(((v.w - mu) * inv) * lw[c + 3] + lb[c + 3]);
}

// ------- kernel 3: QKV GEMM (WMMA, double-buffered TDM pipeline) -------
__global__ __launch_bounds__(256) void qkv_gemm_kernel(const _Float16* __restrict__ h16,
                                                       const _Float16* __restrict__ w16,
                                                       const float* __restrict__ bias,
                                                       _Float16* __restrict__ q16,
                                                       _Float16* __restrict__ k16,
                                                       _Float16* __restrict__ v16) {
    __shared__ __align__(16) _Float16 As[2][128 * 40];   // h tile 128x32, 64B row + 16B pad
    __shared__ __align__(16) _Float16 Bst[2][128 * 40];  // w tile transposed [n][k]
    const int tid  = threadIdx.x;
    const int lane = tid & 31;
    const int wave = tid >> 5;
    const int m0 = blockIdx.y * 128;
    const int n0 = blockIdx.x * 128;
    const int wrow = tid >> 3, wc0 = (tid & 7) * 16;     // W-tile thread coords
#if !HAVE_TDM
    const int arow = tid >> 1, ac0 = (tid & 1) * 16;     // A-tile thread coords
#endif

    // ---- prologue: stage k-tile 0 into buffer 0 ----
#if HAVE_TDM
    if (tid == 0)
        tdm_load_2d_f16(h16 + (size_t)m0 * H_DIM, lds_addr_of(As[0]),
                        32, 128, H_DIM, 32, 128, /*16dw*/ 3, /*4dw*/ 3);
#else
    {
        const uint4* src = (const uint4*)(h16 + (size_t)(m0 + arow) * H_DIM + ac0);
        uint4* dst = (uint4*)(&As[0][arow * 40 + ac0]);
        dst[0] = src[0]; dst[1] = src[1];
    }
#endif
    {
        const _Float16* src = w16 + (size_t)wrow * TH3 + n0 + wc0;
        __align__(16) _Float16 tmp[16];
        *(uint4*)(tmp)     = ((const uint4*)src)[0];
        *(uint4*)(tmp + 8) = ((const uint4*)src)[1];
        #pragma unroll
        for (int j2 = 0; j2 < 16; ++j2) Bst[0][(wc0 + j2) * 40 + wrow] = tmp[j2];
    }
#if HAVE_TDM
    __builtin_amdgcn_s_wait_tensorcnt(0);
#endif
    __syncthreads();

    v8f acc[8] = {};
    for (int it = 0; it < 32; ++it) {
        const int cur = it & 1, nxt = cur ^ 1;
        const int k_next = (it + 1) * 32;
        const bool has_next = (it + 1) < 32;

        // issue next tile's transfers BEFORE computing (overlap with WMMA chain)
        __align__(16) _Float16 wtmp[16];
#if !HAVE_TDM
        uint4 atmp0 = {}, atmp1 = {};
#endif
        if (has_next) {
#if HAVE_TDM
            if (tid == 0)
                tdm_load_2d_f16(h16 + (size_t)m0 * H_DIM + k_next, lds_addr_of(As[nxt]),
                                32, 128, H_DIM, 32, 128, 3, 3);
#else
            const uint4* asrc = (const uint4*)(h16 + (size_t)(m0 + arow) * H_DIM + k_next + ac0);
            atmp0 = asrc[0]; atmp1 = asrc[1];
#endif
            const _Float16* src = w16 + (size_t)(k_next + wrow) * TH3 + n0 + wc0;
            *(uint4*)(wtmp)     = ((const uint4*)src)[0];
            *(uint4*)(wtmp + 8) = ((const uint4*)src)[1];
        }

        // compute on current buffer
        v16h a = load_a_frag(&As[cur][(wave * 16) * 40], 40, lane);
        v16h bfr[8];
        #pragma unroll
        for (int j = 0; j < 8; ++j)
            bfr[j] = load_b_frag(&Bst[cur][(j * 16) * 40], 40, lane);
        #pragma unroll
        for (int j = 0; j < 8; ++j)
            acc[j] = wmma_f16(a, bfr[j], acc[j]);

        // drain next tile's data into LDS after the math
        if (has_next) {
            #pragma unroll
            for (int j2 = 0; j2 < 16; ++j2) Bst[nxt][(wc0 + j2) * 40 + wrow] = wtmp[j2];
#if !HAVE_TDM
            uint4* adst = (uint4*)(&As[nxt][arow * 40 + ac0]);
            adst[0] = atmp0; adst[1] = atmp1;
#endif
        }
#if HAVE_TDM
        __builtin_amdgcn_s_wait_tensorcnt(0);
#endif
        __syncthreads();
    }

    // epilogue: + bias, scatter into [b][head][s][d] f16
    const int m_base = m0 + wave * 16;
    const int col_lo = lane & 15, hi = lane >> 4;
    #pragma unroll
    for (int j = 0; j < 8; ++j) {
        int col = n0 + j * 16 + col_lo;
        float bb = bias[col];
        int part   = col >> 10;
        int within = col & 1023;
        int head   = within >> 6;
        int d      = within & 63;
        _Float16* dst = (part == 0) ? q16 : ((part == 1) ? k16 : v16);
        #pragma unroll
        for (int r = 0; r < 8; ++r) {
            int m = m_base + r + hi * 8;
            int b = m >> 11;
            int s = m & 2047;
            size_t idx = (((size_t)(b * NH + head) * S_LEN + s) * HD) + d;
            dst[idx] = (_Float16)(acc[j][r] + bb);
        }
    }
}

// ---------------- kernel 4: interleaved RoPE on q16, k16 ----------------
__global__ void rope_kernel(const float* __restrict__ sp,
                            _Float16* __restrict__ q16, _Float16* __restrict__ k16) {
    int idx = blockIdx.x * blockDim.x + threadIdx.x;  // B*NH*S*32
    int i  = idx & 31;
    int s  = (idx >> 5) & (S_LEN - 1);
    int bh = idx >> 16;
    float sv = sp[s * 64 + i];        // sin = first half
    float cv = sp[s * 64 + 32 + i];   // cos = second half
    size_t base = (((size_t)bh * S_LEN) + s) * HD + 2 * i;
    float q0 = (float)q16[base], q1 = (float)q16[base + 1];
    q16[base]     = (_Float16)(q0 * cv - q1 * sv);
    q16[base + 1] = (_Float16)(q1 * cv + q0 * sv);
    float kk0 = (float)k16[base], kk1 = (float)k16[base + 1];
    k16[base]     = (_Float16)(kk0 * cv - kk1 * sv);
    k16[base + 1] = (_Float16)(kk1 * cv + kk0 * sv);
}

// ------- kernel 5: flash attention (WMMA, double-buffered TDM pipeline) -------
__global__ __launch_bounds__(256) void attn_kernel(const _Float16* __restrict__ q16,
                                                   const _Float16* __restrict__ k16,
                                                   const _Float16* __restrict__ v16,
                                                   float* __restrict__ out) {
    __shared__ __align__(16) _Float16 Ks[2][64 * 72];  // K tile [kv][d], 128B row + 16B pad
    __shared__ __align__(16) _Float16 Vt[2][64 * 72];  // V tile transposed [d][kv]
    __shared__ __align__(16) _Float16 Ps[128 * 72];    // Q staging, then per-wave P
    const int tid = threadIdx.x, lane = tid & 31, wave = tid >> 5;
    const int bh  = blockIdx.y;
    const int q0g = blockIdx.x * 128;
    const size_t headbase = (size_t)bh * S_LEN * HD;
    const float kScale = 0.125f;   // 1/sqrt(64)
    const int vrow = tid >> 2, vc0 = (tid & 3) * 16;   // K/V-tile thread coords
    const int NT = S_LEN / 64;                          // 32 kv tiles

    // ---- prologue: Q tile + kv tile 0 ----
#if HAVE_TDM
    if (tid == 0) {
        tdm_load_2d_f16(q16 + headbase + (size_t)q0g * HD, lds_addr_of(Ps),
                        64, 128, HD, 64, 128, /*32dw*/ 4, /*4dw*/ 3);
        tdm_load_2d_f16(k16 + headbase, lds_addr_of(Ks[0]),
                        64, 64, HD, 64, 64, 4, 3);
    }
#else
    {
        int row = tid >> 1, c0 = (tid & 1) * 32;
        const uint4* src = (const uint4*)(q16 + headbase + (size_t)(q0g + row) * HD + c0);
        uint4* dst = (uint4*)(Ps + row * 72 + c0);
        dst[0] = src[0]; dst[1] = src[1]; dst[2] = src[2]; dst[3] = src[3];
    }
    {
        const uint4* ksrc = (const uint4*)(k16 + headbase + (size_t)vrow * HD + vc0);
        uint4* kdst = (uint4*)(&Ks[0][vrow * 72 + vc0]);
        kdst[0] = ksrc[0]; kdst[1] = ksrc[1];
    }
#endif
    {   // V tile 0: load + transpose into Vt[0]
        const uint4* vsrc = (const uint4*)(v16 + headbase + (size_t)vrow * HD + vc0);
        __align__(16) _Float16 tmp[16];
        *(uint4*)(tmp)     = vsrc[0];
        *(uint4*)(tmp + 8) = vsrc[1];
        #pragma unroll
        for (int j = 0; j < 16; ++j) Vt[0][(vc0 + j) * 72 + vrow] = tmp[j];
    }
#if HAVE_TDM
    __builtin_amdgcn_s_wait_tensorcnt(0);
#endif
    __syncthreads();
    v16h qa0 = load_a_frag(Ps + (wave * 16) * 72 + 0,  72, lane);
    v16h qa1 = load_a_frag(Ps + (wave * 16) * 72 + 32, 72, lane);
    __syncthreads();   // Ps is reused for P below

    v8f ctx[4] = {};
    float mrow[8], lrow[8];
    #pragma unroll
    for (int r = 0; r < 8; ++r) { mrow[r] = -1e30f; lrow[r] = 0.0f; }

    for (int it = 0; it < NT; ++it) {
        const int cur = it & 1, nxt = cur ^ 1;
        const int kvn = (it + 1) * 64;
        const bool has_next = (it + 1) < NT;

        // issue next tile transfers up-front (overlap with compute below)
        __align__(16) _Float16 vtmp[16];
#if !HAVE_TDM
        uint4 ktmp0 = {}, ktmp1 = {};
#endif
        if (has_next) {
#if HAVE_TDM
            if (tid == 0)
                tdm_load_2d_f16(k16 + headbase + (size_t)kvn * HD, lds_addr_of(Ks[nxt]),
                                64, 64, HD, 64, 64, 4, 3);
#else
            const uint4* ksrc = (const uint4*)(k16 + headbase + (size_t)(kvn + vrow) * HD + vc0);
            ktmp0 = ksrc[0]; ktmp1 = ksrc[1];
#endif
            const uint4* vsrc = (const uint4*)(v16 + headbase + (size_t)(kvn + vrow) * HD + vc0);
            *(uint4*)(vtmp)     = vsrc[0];
            *(uint4*)(vtmp + 8) = vsrc[1];
            if (it + 2 < NT)   // prefetch tile it+2 of the V stream
                __builtin_prefetch(v16 + headbase + (size_t)(kvn + 64 + vrow) * HD + vc0, 0, 3);
        }

        // ---- S = Q * K^T on current tile ----
        v16h kb[8];
        #pragma unroll
        for (int j = 0; j < 4; ++j) {
            kb[2 * j]     = load_b_frag(&Ks[cur][(j * 16) * 72 + 0],  72, lane);
            kb[2 * j + 1] = load_b_frag(&Ks[cur][(j * 16) * 72 + 32], 72, lane);
        }
        v8f sacc[4] = {};
        #pragma unroll
        for (int j = 0; j < 4; ++j) {
            sacc[j] = wmma_f16(qa0, kb[2 * j],     sacc[j]);
            sacc[j] = wmma_f16(qa1, kb[2 * j + 1], sacc[j]);
        }

        // ---- online softmax (row = r + 8*(lane>=16)) ----
        #pragma unroll
        for (int r = 0; r < 8; ++r) {
            float mx = sacc[0][r];
            mx = fmaxf(mx, sacc[1][r]);
            mx = fmaxf(mx, sacc[2][r]);
            mx = fmaxf(mx, sacc[3][r]);
            mx *= kScale;
            #pragma unroll
            for (int msk = 1; msk <= 8; msk <<= 1)
                mx = fmaxf(mx, __shfl_xor(mx, msk, 32));
            float mnew = fmaxf(mrow[r], mx);
            float a = __expf(mrow[r] - mnew);
            mrow[r] = mnew;
            float rs = 0.0f;
            #pragma unroll
            for (int j = 0; j < 4; ++j) {
                float p = __expf(sacc[j][r] * kScale - mnew);
                sacc[j][r] = p;
                rs += p;
            }
            #pragma unroll
            for (int msk = 1; msk <= 8; msk <<= 1)
                rs += __shfl_xor(rs, msk, 32);
            lrow[r] = lrow[r] * a + rs;
            #pragma unroll
            for (int j = 0; j < 4; ++j) ctx[j][r] *= a;
        }

        // ---- P * V on current tile ----
        v16h vb[8];
        #pragma unroll
        for (int j = 0; j < 4; ++j) {
            vb[2 * j]     = load_b_frag(&Vt[cur][(j * 16) * 72 + 0],  72, lane);
            vb[2 * j + 1] = load_b_frag(&Vt[cur][(j * 16) * 72 + 32], 72, lane);
        }
        {   // stage P in wave-private LDS (same-wave DS ops are in-order)
            _Float16* prow = Ps + (wave * 16) * 72;
            int cl = lane & 15, hi = lane >> 4;
            #pragma unroll
            for (int j = 0; j < 4; ++j) {
                #pragma unroll
                for (int r = 0; r < 8; ++r)
                    prow[(r + 8 * hi) * 72 + j * 16 + cl] = (_Float16)sacc[j][r];
            }
        }
        v16h pa0 = load_a_frag(Ps + (wave * 16) * 72 + 0,  72, lane);
        v16h pa1 = load_a_frag(Ps + (wave * 16) * 72 + 32, 72, lane);
        #pragma unroll
        for (int j = 0; j < 4; ++j) {
            ctx[j] = wmma_f16(pa0, vb[2 * j],     ctx[j]);
            ctx[j] = wmma_f16(pa1, vb[2 * j + 1], ctx[j]);
        }

        // drain next tile into LDS after the math
        if (has_next) {
            #pragma unroll
            for (int j = 0; j < 16; ++j) Vt[nxt][(vc0 + j) * 72 + vrow] = vtmp[j];
#if !HAVE_TDM
            uint4* kdst = (uint4*)(&Ks[nxt][vrow * 72 + vc0]);
            kdst[0] = ktmp0; kdst[1] = ktmp1;
#endif
        }
#if HAVE_TDM
        __builtin_amdgcn_s_wait_tensorcnt(0);
#endif
        __syncthreads();
    }

    // epilogue: normalize and write fp32 out [b][s][head*64 + d]
    const int b = bh >> 4, head = bh & 15;
    const int cl = lane & 15, hi = lane >> 4;
    #pragma unroll
    for (int r = 0; r < 8; ++r) {
        float invl = 1.0f / lrow[r];
        int s = q0g + wave * 16 + r + 8 * hi;
        float* orow = out + ((size_t)(b * S_LEN + s)) * H_DIM + head * HD;
        #pragma unroll
        for (int j = 0; j < 4; ++j)
            orow[j * 16 + cl] = ctx[j][r] * invl;
    }
}

extern "C" void kernel_launch(void* const* d_in, const int* in_sizes, int n_in,
                              void* d_out, int out_size, void* d_ws, size_t ws_size,
                              hipStream_t stream) {
    (void)in_sizes; (void)n_in; (void)out_size; (void)ws_size;
    const float* hs = (const float*)d_in[0];   // hidden_states [4,2048,1024]
    const float* sp = (const float*)d_in[1];   // sinusoidal_pos [1,1,2048,64]
    const float* lw = (const float*)d_in[2];   // ln_weight [1024]
    const float* lb = (const float*)d_in[3];   // ln_bias [1024]
    const float* wq = (const float*)d_in[4];   // w_qkv [1024,3072]
    const float* bq = (const float*)d_in[5];   // b_qkv [3072]
    float* out = (float*)d_out;

    char* ws = (char*)d_ws;
    _Float16* w16 = (_Float16*)(ws + 0);           //  6,291,456 B
    _Float16* h16 = (_Float16*)(ws + 6291456);     // 16,777,216 B
    _Float16* q16 = (_Float16*)(ws + 23068672);    // 16,777,216 B
    _Float16* k16 = (_Float16*)(ws + 39845888);    // 16,777,216 B
    _Float16* v16 = (_Float16*)(ws + 56623104);    // 16,777,216 B

    cvt_w_kernel<<<3072, 256, 0, stream>>>(wq, w16);
    ln_kernel<<<BS, 256, 0, stream>>>(hs, lw, lb, h16);
    dim3 gg(TH3 / 128, BS / 128);                  // 24 x 64
    qkv_gemm_kernel<<<gg, 256, 0, stream>>>(h16, w16, bq, q16, k16, v16);
    rope_kernel<<<16384, 256, 0, stream>>>(sp, q16, k16);
    dim3 ga(S_LEN / 128, BATCH * NH);              // 16 x 64
    attn_kernel<<<ga, 256, 0, stream>>>(q16, k16, v16, out);
}
